// AttentionPooling_86268713107582
// MI455X (gfx1250) — compile-verified, matrix operand order checked by reading, not run on hardware
//
#include <hip/hip_runtime.h>

typedef _Float16 f16;
typedef __attribute__((ext_vector_type(16))) _Float16 v16h;
typedef __attribute__((ext_vector_type(8)))  _Float16 v8h;
typedef __attribute__((ext_vector_type(8)))  float    v8f;
typedef __attribute__((ext_vector_type(4)))  float    v4f;

constexpr int Bb = 8;     // batch
constexpr int Nn = 4096;  // sequence
constexpr int Dd = 256;   // IN_DIM
constexpr int Hh = 256;   // HID

// Load a 16x32 f16 fragment slice for this lane: 8 contiguous halfs at p
// (K = g*8 + 0..7) and 8 at p+16 halfs (K = 16 + g*8 + 0..7).
__device__ __forceinline__ v16h ld_frag(const f16* p) {
  v8h lo = *(const v8h*)p;
  v8h hi = *(const v8h*)(p + 16);
  v16h r;
#pragma unroll
  for (int i = 0; i < 8; ++i) { r[i] = lo[i]; r[i + 8] = hi[i]; }
  return r;
}

__device__ __forceinline__ v8f wmma16(v16h a, v16h b, v8f c) {
  return __builtin_amdgcn_wmma_f32_16x16x32_f16(false, a, false, b, (short)0, c,
                                                false, false);
}

// Async copy 16 bytes global -> LDS (GLOBAL_LOAD_ASYNC_TO_LDS_B128, ASYNCcnt).
// Generic shared-pointer low 32 bits == LDS byte offset (aperture rule).
__device__ __forceinline__ void async_b128(const void* gsrc, const void* ldst) {
  unsigned l = (unsigned)(uintptr_t)ldst;
  unsigned long long g = (unsigned long long)(uintptr_t)gsrc;
  asm volatile("global_load_async_to_lds_b128 %0, %1, off"
               :: "v"(l), "v"(g) : "memory");
}
__device__ __forceinline__ void wait_async0() {
  asm volatile("s_wait_asynccnt 0x0" ::: "memory");
}

// ---------------------------------------------------------------------------
// Phase 0: pack Wq/Wk/Wv (f32 row-major) into f16 B-fragment lane order.
// Fragment (m, t, c): lane L (g=L>>4, nl=L&15) gets 16 halfs:
//   element i -> W[c*32 + (i<8?0:16) + g*8 + (i&7)][t*16 + nl]
// Packed flat index = (((m*16+t)*8 + c)*32 + L)*16 + i  (32B per lane).
// ---------------------------------------------------------------------------
__global__ __launch_bounds__(256) void pack_w_kernel(
    const float* __restrict__ Wq, const float* __restrict__ Wk,
    const float* __restrict__ Wv, f16* __restrict__ Wp) {
  const int id = blockIdx.x * 256 + threadIdx.x;  // 0 .. 3*16*8*32-1
  const int L  = id & 31;
  const int c  = (id >> 5) & 7;
  const int t  = (id >> 8) & 15;
  const int m  = id >> 12;
  const float* W = (m == 0) ? Wq : ((m == 1) ? Wk : Wv);
  const int g = L >> 4, nl = L & 15;
  v16h fr;
#pragma unroll
  for (int i = 0; i < 16; ++i) {
    const int k = c * 32 + ((i < 8) ? 0 : 16) + g * 8 + (i & 7);
    fr[i] = (f16)W[(size_t)k * Hh + t * 16 + nl];
  }
  *(v16h*)(Wp + (size_t)id * 16) = fr;
}

// ---------------------------------------------------------------------------
// Phase 1: Q = xWq+bq, K = xWk+bk (row-major f16), V = xWv+bv stored
// transposed Vt[b][h][n] (f16). One wave per 16-row tile of x.
// ---------------------------------------------------------------------------
__global__ __launch_bounds__(32) void proj_kernel(
    const float* __restrict__ x, const f16* __restrict__ Wp,
    const float* __restrict__ bq, const float* __restrict__ bk,
    const float* __restrict__ bv,
    f16* __restrict__ Qh, f16* __restrict__ Kh, f16* __restrict__ Vt) {
  const int lane = threadIdx.x & 31;
  const int g    = lane >> 4;
  const int nl   = lane & 15;
  const int bid  = blockIdx.x;
  const int b    = bid / (Nn / 16);
  const int row0 = (bid % (Nn / 16)) * 16;

  // x fragments for this 16-row tile: 8 chunks of K=32 over IN_DIM=256
  v16h xa[8];
  const float* xrow = x + ((size_t)(b * Nn + row0 + nl)) * Dd;
#pragma unroll
  for (int c = 0; c < 8; ++c) {
    const float* p = xrow + c * 32 + g * 8;
    const v4f f0 = *(const v4f*)(p);
    const v4f f1 = *(const v4f*)(p + 4);
    const v4f f2 = *(const v4f*)(p + 16);
    const v4f f3 = *(const v4f*)(p + 20);
    v16h a;
#pragma unroll
    for (int i = 0; i < 4; ++i) {
      a[i]      = (f16)f0[i];  a[i + 4]  = (f16)f1[i];
      a[i + 8]  = (f16)f2[i];  a[i + 12] = (f16)f3[i];
    }
    xa[c] = a;
  }

#pragma unroll
  for (int m = 0; m < 3; ++m) {
    const float* bias = (m == 0) ? bq : ((m == 1) ? bk : bv);
#pragma unroll 1
    for (int t = 0; t < 16; ++t) {       // output column tile
      v8f cacc = {};
      const f16* wp = Wp + (((size_t)(m * 16 + t) * 8) * 32 + lane) * 16;
#pragma unroll
      for (int c = 0; c < 8; ++c)
        cacc = wmma16(xa[c], *(const v16h*)(wp + (size_t)c * 32 * 16), cacc);
      const float bv_ = bias[t * 16 + nl];
#pragma unroll
      for (int r = 0; r < 8; ++r) {
        const int grow = row0 + r + 8 * g;   // token
        const int gcol = t * 16 + nl;        // feature
        const f16 hv   = (f16)(cacc[r] + bv_);
        if (m == 0)      Qh[((size_t)(b * Nn + grow)) * Dd + gcol] = hv;
        else if (m == 1) Kh[((size_t)(b * Nn + grow)) * Dd + gcol] = hv;
        else             Vt[((size_t)(b * Dd + gcol)) * Nn + grow] = hv;
      }
    }
  }
}

// ---------------------------------------------------------------------------
// Phase 2: flash attention. 128 threads = 4 waves; each wave owns a 16-row
// q-tile. K (32x256) and Vt (256x32) tiles staged via async-to-LDS; next
// tile prefetched into L2 while WMMAs run. Online softmax in f32.
// ---------------------------------------------------------------------------
__global__ __launch_bounds__(128) void attn_kernel(
    const f16* __restrict__ Qh, const f16* __restrict__ Kh,
    const f16* __restrict__ Vt, float* __restrict__ partials) {
  __shared__ __align__(16) f16 Ks[32 * 256];      // [key][d]   16 KB
  __shared__ __align__(16) f16 Vs[256 * 32];      // [h][key]   16 KB
  __shared__ __align__(16) f16 Ps[4][16 * 32];    // per-wave P  4 KB

  const int tid  = threadIdx.x;
  const int lane = tid & 31;
  const int wave = tid >> 5;
  const int g    = lane >> 4;
  const int nl   = lane & 15;
  const int b    = blockIdx.x / (Nn / 64);
  const int qblk = blockIdx.x % (Nn / 64);
  const int qrow = qblk * 64 + wave * 16 + nl;

  // Q fragments for this wave's 16 query rows (full D=256 -> 8 chunks)
  v16h qa[8];
  const f16* qrp = Qh + ((size_t)(b * Nn + qrow)) * Dd;
#pragma unroll
  for (int c = 0; c < 8; ++c) qa[c] = ld_frag(qrp + c * 32 + g * 8);

  float mi[8], li[8];
  v8f acc[16];
#pragma unroll
  for (int r = 0; r < 8; ++r) { mi[r] = -3.0e38f; li[r] = 0.f; }
#pragma unroll
  for (int t = 0; t < 16; ++t) acc[t] = (v8f){};

  const float scale = 0.0625f;  // 1/sqrt(256)
  const f16* Kbase = Kh + (size_t)b * Nn * Dd;
  const f16* Vbase = Vt + (size_t)b * Dd * Nn;

#pragma unroll 1
  for (int kt = 0; kt < Nn / 32; ++kt) {
    const int key0 = kt * 32;

    // ---- async staging (GLOBAL_LOAD_ASYNC_TO_LDS_B128, no VGPR data) ----
#pragma unroll
    for (int j = 0; j < 8; ++j) {          // K: 1024 x 16B units
      const int u = tid + 128 * j;
      async_b128(Kbase + (size_t)key0 * Dd + u * 8, Ks + u * 8);
    }
#pragma unroll
    for (int j = 0; j < 8; ++j) {          // Vt: rows h, 4 units per row
      const int u = tid + 128 * j;
      const int h = u >> 2, part = u & 3;
      async_b128(Vbase + (size_t)h * Nn + key0 + part * 8,
                 Vs + h * 32 + part * 8);
    }
    wait_async0();
    __syncthreads();

    // ---- prefetch next tile into L2 while this tile computes ----
    if (kt + 1 < Nn / 32) {
      const int key1 = key0 + 32;
      __builtin_prefetch(Kbase + (size_t)key1 * Dd + tid * 64, 0, 1);
      __builtin_prefetch(Vbase + (size_t)(2 * tid) * Nn + key1, 0, 1);
      __builtin_prefetch(Vbase + (size_t)(2 * tid + 1) * Nn + key1, 0, 1);
    }

    // ---- S = Q·Kt for two 16-key tiles ----
    v8f s0 = {}, s1 = {};
#pragma unroll
    for (int c = 0; c < 8; ++c)
      s0 = wmma16(qa[c], ld_frag(Ks + (size_t)nl * Dd + c * 32 + g * 8), s0);
#pragma unroll
    for (int c = 0; c < 8; ++c)
      s1 = wmma16(qa[c], ld_frag(Ks + (size_t)(nl + 16) * Dd + c * 32 + g * 8), s1);

    // ---- online softmax (per-row stats; rows r+8g, cols across 16 lanes) ----
    float alpha[8];
    v8f p0 = {}, p1 = {};
#pragma unroll
    for (int r = 0; r < 8; ++r) {
      const float a0 = s0[r] * scale, a1 = s1[r] * scale;
      float rm = fmaxf(a0, a1);
      rm = fmaxf(rm, __shfl_xor(rm, 1, 32));
      rm = fmaxf(rm, __shfl_xor(rm, 2, 32));
      rm = fmaxf(rm, __shfl_xor(rm, 4, 32));
      rm = fmaxf(rm, __shfl_xor(rm, 8, 32));
      const float mn = fmaxf(mi[r], rm);
      const float al = __expf(mi[r] - mn);
      const float e0 = __expf(a0 - mn), e1 = __expf(a1 - mn);
      float rs = e0 + e1;
      rs += __shfl_xor(rs, 1, 32);
      rs += __shfl_xor(rs, 2, 32);
      rs += __shfl_xor(rs, 4, 32);
      rs += __shfl_xor(rs, 8, 32);
      li[r] = li[r] * al + rs;
      mi[r] = mn;
      alpha[r] = al;
      p0[r] = e0; p1[r] = e1;
    }
#pragma unroll
    for (int t = 0; t < 16; ++t)
#pragma unroll
      for (int r = 0; r < 8; ++r) acc[t][r] *= alpha[r];

    // ---- C-layout P -> A-layout via per-wave LDS (same-wave DS in-order) --
    f16* pw = &Ps[wave][0];
#pragma unroll
    for (int r = 0; r < 8; ++r) {
      pw[(r + 8 * g) * 32 + nl]      = (f16)p0[r];
      pw[(r + 8 * g) * 32 + 16 + nl] = (f16)p1[r];
    }
    const v16h pa = ld_frag(pw + nl * 32 + g * 8);

    // ---- O += P·V over 16 h-tiles ----
#pragma unroll
    for (int t = 0; t < 16; ++t)
      acc[t] = wmma16(pa, ld_frag(Vs + (size_t)(t * 16 + nl) * 32 + g * 8), acc[t]);

    __syncthreads();
  }

  // ---- normalize by row sums, reduce this wave's 16 query rows ----
#pragma unroll
  for (int t = 0; t < 16; ++t) {
    float ps = 0.f;
#pragma unroll
    for (int r = 0; r < 8; ++r) ps += acc[t][r] / li[r];
    ps += __shfl_xor(ps, 16, 32);    // add the other 8 rows (same column)
    if (g == 0) {
      const int tq = qblk * 4 + wave;
      partials[((size_t)(b * (Nn / 16) + tq)) * Hh + t * 16 + nl] = ps;
    }
  }
}

// ---------------------------------------------------------------------------
// Phase 3: deterministic reduction of q-tile partials -> mean over N queries.
// ---------------------------------------------------------------------------
__global__ __launch_bounds__(256) void reduce_kernel(
    const float* __restrict__ partials, float* __restrict__ out) {
  const int b = blockIdx.x;
  const int h = threadIdx.x;
  float s = 0.f;
#pragma unroll 1
  for (int t = 0; t < Nn / 16; ++t)
    s += partials[((size_t)(b * (Nn / 16) + t)) * Hh + h];
  out[b * Hh + h] = s * (1.0f / (float)Nn);
}

extern "C" void kernel_launch(void* const* d_in, const int* in_sizes, int n_in,
                              void* d_out, int out_size, void* d_ws,
                              size_t ws_size, hipStream_t stream) {
  const float* x  = (const float*)d_in[0];
  const float* Wq = (const float*)d_in[1];
  const float* bq = (const float*)d_in[2];
  const float* Wk = (const float*)d_in[3];
  const float* bk = (const float*)d_in[4];
  const float* Wv = (const float*)d_in[5];
  const float* bv = (const float*)d_in[6];
  float* out = (float*)d_out;

  // Workspace (~50.4 MB): Qh | Kh | Vt (f16, 16 MB each) | partials | packed W
  char* ws = (char*)d_ws;
  const size_t mat = (size_t)Bb * Nn * Dd * sizeof(f16);
  f16* Qh = (f16*)(ws);
  f16* Kh = (f16*)(ws + mat);
  f16* Vt = (f16*)(ws + 2 * mat);
  float* partials = (float*)(ws + 3 * mat);
  f16* Wp = (f16*)(ws + 3 * mat + (size_t)Bb * (Nn / 16) * Hh * sizeof(float));

  pack_w_kernel<<<(3 * 16 * 8 * 32) / 256, 256, 0, stream>>>(Wq, Wk, Wv, Wp);
  proj_kernel<<<Bb * (Nn / 16), 32, 0, stream>>>(x, Wp, bq, bk, bv, Qh, Kh, Vt);
  attn_kernel<<<Bb * (Nn / 64), 128, 0, stream>>>(Qh, Kh, Vt, partials);
  reduce_kernel<<<Bb, Hh, 0, stream>>>(partials, out);
}
